// Tacotron2Decoder_18468359373410
// MI455X (gfx1250) — matrix operand order repaired and checked by reading, not compiled, not run on hardware
//
#include <hip/hip_runtime.h>
#include <stdint.h>

// ---------------- problem constants ----------------
#define B_      64
#define T_      800
#define ENC_    512
#define SPK_    256
#define ODIM_   80
#define ODIMP_  96              // ODIM zero-padded to a K-multiple of 32
#define PRE_    256
#define H_      1024
#define IN0_    1024            // ENC+SPK+PRE
#define FEATIN_ 1792            // H+ENC+SPK
#define PCH_    512
#define NPAD_   256             // N padded to one block tile for N=80 GEMMs
#define BT_     (B_ * T_)       // 51200 rows, t-major: row = t*B_ + b
#define TP_     (T_ + 4)        // time-padded rows for conv (pad 2 each side)
#define NBLK_   32              // persistent LSTM grid (must all be resident)

typedef unsigned short u16t;
typedef __attribute__((ext_vector_type(16))) __bf16 v16bf;
typedef __attribute__((ext_vector_type(8)))  float  v8f;

// ---------------- scalar helpers ----------------
__device__ __forceinline__ u16t f2bf(float f) {
  union { float f; unsigned u; } x; x.f = f;
  unsigned u = x.u;
  u += 0x7FFFu + ((u >> 16) & 1u);          // round-to-nearest-even
  return (u16t)(u >> 16);
}
__device__ __forceinline__ float bf2f(u16t h) {
  union { unsigned u; float f; } x; x.u = ((unsigned)h) << 16; return x.f;
}
__device__ __forceinline__ float sigm(float x) { return 1.f / (1.f + __expf(-x)); }

union ABf { u16t u[16]; v16bf v; };

// Branch-free fragment loads (K must be a multiple of 32; buffers are pre-padded).
// A fragment (16x32 bf16, MxK), ISA 7.12.2 layout:
//   half h: elems 0..7 <- K = k0+8h.. ; elems 8..15 <- K = k0+16+8h..
__device__ __forceinline__ void load_a_frag(const u16t* __restrict__ A, int lda,
                                            int row, int k0, ABf& a) {
  const int half = (threadIdx.x >> 4) & 1;
  const u16t* p = A + (size_t)row * lda + k0 + 8 * half;
  *(uint4*)&a.u[0] = *(const uint4*)(p);
  *(uint4*)&a.u[8] = *(const uint4*)(p + 16);
}
// B fragment (32x16 bf16, KxN) from weights stored row-major W[N][K] (B = W^T):
//   lane = n%16; half h: elems 0..7 <- K = k0+8h.. ; elems 8..15 <- K = k0+16+8h..
__device__ __forceinline__ void load_b_frag(const u16t* __restrict__ W, int ldw,
                                            int n, int k0, ABf& b) {
  const int half = (threadIdx.x >> 4) & 1;
  const u16t* p = W + (size_t)n * ldw + k0 + 8 * half;
  *(uint4*)&b.u[0] = *(const uint4*)(p);
  *(uint4*)&b.u[8] = *(const uint4*)(p + 16);
}

// One K=32 step of a 16(M) x 64(N) wave tile: issue ALL fragment loads first
// (clauseable, one wait), then 4 back-to-back WMMAs sharing the A fragment.
__device__ __forceinline__ void gemm_step(const u16t* __restrict__ A, int lda, int arow,
                                          const u16t* __restrict__ W, int ldw,
                                          int n0, int ln, int k0, v8f* acc) {
  ABf a;  load_a_frag(A, lda, arow, k0, a);
  ABf b0; load_b_frag(W, ldw, n0 +  0 + ln, k0, b0);
  ABf b1; load_b_frag(W, ldw, n0 + 16 + ln, k0, b1);
  ABf b2; load_b_frag(W, ldw, n0 + 32 + ln, k0, b2);
  ABf b3; load_b_frag(W, ldw, n0 + 48 + ln, k0, b3);
  acc[0] = __builtin_amdgcn_wmma_f32_16x16x32_bf16(false, a.v, false, b0.v, (short)0, acc[0], false, false);
  acc[1] = __builtin_amdgcn_wmma_f32_16x16x32_bf16(false, a.v, false, b1.v, (short)0, acc[1], false, false);
  acc[2] = __builtin_amdgcn_wmma_f32_16x16x32_bf16(false, a.v, false, b2.v, (short)0, acc[2], false, false);
  acc[3] = __builtin_amdgcn_wmma_f32_16x16x32_bf16(false, a.v, false, b3.v, (short)0, acc[3], false, false);
}

// ---------------- generic bf16 WMMA GEMM:  out = act(A[MxK] * W[NxK]^T + bias) ----------------
// block = 256 thr = 8 waves; wave tile = 16(M) x 64(N); block tile = 32(M) x 256(N).
// W must have >= gridDim.x*256 rows (zero-padded); K % 32 == 0. N only masks the epilogue.
__global__ __launch_bounds__(256) void k_gemm(
    const u16t* __restrict__ A, int lda,
    const u16t* __restrict__ W, int ldw,
    const float* __restrict__ bias,
    float* __restrict__ outF, int ldof,
    u16t*  __restrict__ outB, int ldob,
    int M, int N, int K, int act) {
  const int lane = threadIdx.x & 31;
  const int ln = lane & 15, half = lane >> 4;
  const int wv = threadIdx.x >> 5;
  const int row0 = blockIdx.y * 32 + (wv >> 2) * 16;
  const int n0   = blockIdx.x * 256 + (wv & 3) * 64;
  if (row0 >= M) return;

  v8f acc[4];
#pragma unroll
  for (int j = 0; j < 4; j++)
#pragma unroll
    for (int r = 0; r < 8; r++) acc[j][r] = 0.f;

  for (int k0 = 0; k0 < K; k0 += 32)
    gemm_step(A, lda, row0 + ln, W, ldw, n0, ln, k0, acc);

#pragma unroll
  for (int j = 0; j < 4; j++) {
    const int nc = n0 + j * 16 + ln;
    if (nc >= N) continue;
    const float bv = bias ? bias[nc] : 0.f;
#pragma unroll
    for (int r = 0; r < 8; r++) {
      const int rr = row0 + r + 8 * half;
      float v = acc[j][r] + bv;
      if (act == 1) v = v > 0.f ? v : 0.f;
      else if (act == 2) v = tanhf(v);
      if (outF) outF[(size_t)rr * ldof + nc] = v;
      if (outB) outB[(size_t)rr * ldob + nc] = f2bf(v);
    }
  }
}

// ---------------- conv1d (K=5, SAME) as 5 shifted GEMMs, channels-last, t-major ----------------
// Apad: padded activations (B*(T+4) rows x Cin), tap dk reads row (r + dk*B_).
// Wp: weights repacked (5, CoutW, Cin) bf16, CoutW >= gridDim.x*256 (padded rows zero).
__global__ __launch_bounds__(256) void k_conv(
    const u16t* __restrict__ Apad,
    const u16t* __restrict__ Wp,
    const float* __restrict__ Cadd,    // residual (rows t-major, ld = Cout), may be null
    float* __restrict__ outF,          // may be null
    u16t*  __restrict__ outB,          // may be null (pre-offset to interior rows)
    int Cout, int CoutW, int Cin, int act) {
  const int lane = threadIdx.x & 31;
  const int ln = lane & 15, half = lane >> 4;
  const int wv = threadIdx.x >> 5;
  const int row0 = blockIdx.y * 32 + (wv >> 2) * 16;
  const int n0   = blockIdx.x * 256 + (wv & 3) * 64;

  v8f acc[4];
#pragma unroll
  for (int j = 0; j < 4; j++) {
    const int nc = n0 + j * 16 + ln;
#pragma unroll
    for (int r = 0; r < 8; r++) {
      const int rr = row0 + r + 8 * half;
      acc[j][r] = (Cadd && nc < Cout) ? Cadd[(size_t)rr * Cout + nc] : 0.f;
    }
  }
#pragma unroll
  for (int dk = 0; dk < 5; dk++) {
    const u16t* At = Apad + (size_t)dk * B_ * Cin;
    const u16t* Wt = Wp + (size_t)dk * CoutW * Cin;
    for (int k0 = 0; k0 < Cin; k0 += 32)
      gemm_step(At, Cin, row0 + ln, Wt, Cin, n0, ln, k0, acc);
  }
#pragma unroll
  for (int j = 0; j < 4; j++) {
    const int nc = n0 + j * 16 + ln;
    if (nc >= Cout) continue;
#pragma unroll
    for (int r = 0; r < 8; r++) {
      const int rr = row0 + r + 8 * half;
      float v = acc[j][r];
      if (act == 2) v = tanhf(v);
      if (outF) outF[(size_t)rr * Cout + nc] = v;
      if (outB) outB[(size_t)rr * Cout + nc] = f2bf(v);
    }
  }
}

// ---------------- grid-wide sense barrier for the persistent LSTM kernel ----------------
__device__ __forceinline__ void gridbar(unsigned* bar) {
  __threadfence();
  __syncthreads();
  if (threadIdx.x == 0) {
    const unsigned gen = atomicAdd(&bar[1], 0u);
    if (atomicAdd(&bar[0], 1u) == (unsigned)(NBLK_ - 1)) {
      atomicExch(&bar[0], 0u);
      __threadfence();
      atomicAdd(&bar[1], 1u);
    } else {
      while (atomicAdd(&bar[1], 0u) == gen) { __builtin_amdgcn_s_sleep(8); }
    }
  }
  __syncthreads();
}

// ---------------- persistent 2-layer LSTM over T steps ----------------
// 32 blocks x 8 waves = 256 waves = exact tiling of the 64x4096 gate GEMM
// (4 M-tiles of 16 x 64 N-tiles of 64). g0 holds the precomputed input projection + biases.
__global__ __launch_bounds__(256) void k_lstm(
    const u16t* __restrict__ g0,                          // (T*B, 4H) bf16, t-major
    const u16t* __restrict__ Whh0, const u16t* __restrict__ Wih1,
    const u16t* __restrict__ Whh1,
    const float* __restrict__ bsum1,                      // b_ih1+b_hh1 (4H)
    u16t* __restrict__ zcs,                               // (T*B, FEATIN) bf16; write cols [0,H)
    float* __restrict__ gbuf0, float* __restrict__ gbuf1, // (B, 4H) f32
    float* __restrict__ c0s, float* __restrict__ c1s,     // (B, H) f32
    u16t*  __restrict__ h0s, u16t*  __restrict__ h1s,     // (B, H) bf16
    unsigned* __restrict__ bar) {
  const int lane = threadIdx.x & 31;
  const int ln = lane & 15, half = lane >> 4;
  const int wv = threadIdx.x >> 5;
  const int g = blockIdx.x * 8 + wv;     // 0..255
  const int n0   = (g >> 2) * 64;        // N tile base (4H = 4096)
  const int row0 = (g & 3) * 16;         // M tile base (B = 64)

  for (int t = 0; t < T_; ++t) {
    // ---- phase A: gates0 = g0[t] + h0 @ Whh0^T
    v8f acc[4];
    const u16t* g0row = g0 + (size_t)(t * B_) * (4 * H_);
#pragma unroll
    for (int j = 0; j < 4; j++) {
      const int nc = n0 + j * 16 + ln;
#pragma unroll
      for (int r = 0; r < 8; r++)
        acc[j][r] = bf2f(g0row[(size_t)(row0 + r + 8 * half) * (4 * H_) + nc]);
    }
    for (int k0 = 0; k0 < H_; k0 += 32)
      gemm_step(h0s, H_, row0 + ln, Whh0, H_, n0, ln, k0, acc);
#pragma unroll
    for (int j = 0; j < 4; j++) {
      const int nc = n0 + j * 16 + ln;
#pragma unroll
      for (int r = 0; r < 8; r++)
        gbuf0[(size_t)(row0 + r + 8 * half) * (4 * H_) + nc] = acc[j][r];
    }
    gridbar(bar);

    // ---- phase B: layer-0 pointwise (i,f,g,o)
    for (int e = blockIdx.x * 256 + threadIdx.x; e < B_ * H_; e += NBLK_ * 256) {
      const int b = e >> 10, u = e & (H_ - 1);
      const float* gb = gbuf0 + (size_t)b * (4 * H_);
      const float cc = sigm(gb[u + H_]) * c0s[e] + sigm(gb[u]) * tanhf(gb[u + 2 * H_]);
      c0s[e] = cc;
      h0s[e] = f2bf(sigm(gb[u + 3 * H_]) * tanhf(cc));
    }
    gridbar(bar);

    // ---- phase C: gates1 = bsum1 + h0 @ Wih1^T + h1 @ Whh1^T
#pragma unroll
    for (int j = 0; j < 4; j++) {
      const float bv = bsum1[n0 + j * 16 + ln];
#pragma unroll
      for (int r = 0; r < 8; r++) acc[j][r] = bv;
    }
    for (int k0 = 0; k0 < H_; k0 += 32)
      gemm_step(h0s, H_, row0 + ln, Wih1, H_, n0, ln, k0, acc);
    for (int k0 = 0; k0 < H_; k0 += 32)
      gemm_step(h1s, H_, row0 + ln, Whh1, H_, n0, ln, k0, acc);
#pragma unroll
    for (int j = 0; j < 4; j++) {
      const int nc = n0 + j * 16 + ln;
#pragma unroll
      for (int r = 0; r < 8; r++)
        gbuf1[(size_t)(row0 + r + 8 * half) * (4 * H_) + nc] = acc[j][r];
    }
    gridbar(bar);

    // ---- phase D: layer-1 pointwise, emit z[t]=h1 into zcs
    for (int e = blockIdx.x * 256 + threadIdx.x; e < B_ * H_; e += NBLK_ * 256) {
      const int b = e >> 10, u = e & (H_ - 1);
      const float* gb = gbuf1 + (size_t)b * (4 * H_);
      const float cc = sigm(gb[u + H_]) * c1s[e] + sigm(gb[u]) * tanhf(gb[u + 2 * H_]);
      c1s[e] = cc;
      const u16t hv = f2bf(sigm(gb[u + 3 * H_]) * tanhf(cc));
      h1s[e] = hv;
      zcs[(size_t)(t * B_ + b) * FEATIN_ + u] = hv;
    }
    gridbar(bar);
  }
}

// ---------------- small elementwise kernels ----------------
__global__ void k_zero(unsigned* __restrict__ p, size_t n) {
  size_t i = (size_t)blockIdx.x * blockDim.x + threadIdx.x;
  const size_t st = (size_t)gridDim.x * blockDim.x;
  for (; i < n; i += st) p[i] = 0u;
}
__global__ void k_cvt(const float* __restrict__ in, u16t* __restrict__ out, size_t n) {
  size_t i = (size_t)blockIdx.x * blockDim.x + threadIdx.x;
  const size_t st = (size_t)gridDim.x * blockDim.x;
  for (; i < n; i += st) out[i] = f2bf(in[i]);
}
// f32 (Nin,Kin) -> bf16 (Npad,Kpad) with zero padding
__global__ void k_cvt_pad(const float* __restrict__ in, u16t* __restrict__ out,
                          int Nin, int Kin, int Npad, int Kpad) {
  size_t i = (size_t)blockIdx.x * blockDim.x + threadIdx.x;
  const size_t st = (size_t)gridDim.x * blockDim.x;
  const size_t n = (size_t)Npad * Kpad;
  for (; i < n; i += st) {
    const int k = (int)(i % Kpad);
    const int nn = (int)(i / Kpad);
    const float v = (nn < Nin && k < Kin) ? in[(size_t)nn * Kin + k] : 0.f;
    out[i] = f2bf(v);
  }
}
__global__ void k_add(const float* __restrict__ a, const float* __restrict__ b,
                      float* __restrict__ o, int n) {
  int i = blockIdx.x * blockDim.x + threadIdx.x;
  if (i < n) o[i] = a[i] + b[i];
}
// teacher-forced prenet input: row (t,b) <- target[b, t-1] (zeros at t==0), t-major bf16,
// K padded to ODIMP_ with zeros
__global__ void k_pack_prev(const float* __restrict__ tgt, u16t* __restrict__ out) {
  size_t i = (size_t)blockIdx.x * blockDim.x + threadIdx.x;
  const size_t st = (size_t)gridDim.x * blockDim.x;
  const size_t n = (size_t)BT_ * ODIMP_;
  for (; i < n; i += st) {
    const int c = (int)(i % ODIMP_);
    const int r = (int)(i / ODIMP_);
    const int t = r / B_, b = r % B_;
    const float v = (t == 0 || c >= ODIM_) ? 0.f
                  : tgt[((size_t)b * T_ + (t - 1)) * ODIM_ + c];
    out[i] = f2bf(v);
  }
}
// scatter h_t and speaker embedding into xs (for layer-0 proj) and zcs (for feat_out)
__global__ void k_pack_xz(const float* __restrict__ ht, const float* __restrict__ spk,
                          u16t* __restrict__ xs, u16t* __restrict__ zcs) {
  size_t i = (size_t)blockIdx.x * blockDim.x + threadIdx.x;
  const size_t st = (size_t)gridDim.x * blockDim.x;
  const size_t n = (size_t)BT_ * (ENC_ + SPK_);
  for (; i < n; i += st) {
    const int c = (int)(i % (ENC_ + SPK_));
    const int r = (int)(i / (ENC_ + SPK_));
    const int t = r / B_, b = r % B_;
    if (c < ENC_) {
      const u16t v = f2bf(ht[((size_t)b * T_ + t) * ENC_ + c]);
      xs[(size_t)r * IN0_ + c] = v;
      zcs[(size_t)r * FEATIN_ + H_ + c] = v;
    } else {
      const int cs = c - ENC_;
      const u16t v = f2bf(spk[(size_t)b * SPK_ + cs]);
      xs[(size_t)r * IN0_ + ENC_ + PRE_ + cs] = v;
      zcs[(size_t)r * FEATIN_ + H_ + ENC_ + cs] = v;
    }
  }
}
// repack conv weight (Co,Ci,5) f32 -> (5,Cop,Cip) bf16 with zero padding
__global__ void k_convw(const float* __restrict__ w, u16t* __restrict__ out,
                        int Co, int Ci, int Cop, int Cip) {
  size_t i = (size_t)blockIdx.x * blockDim.x + threadIdx.x;
  const size_t st = (size_t)gridDim.x * blockDim.x;
  const size_t n = (size_t)5 * Cop * Cip;
  for (; i < n; i += st) {
    const int dk = (int)(i / ((size_t)Cop * Cip));
    const size_t rem = i % ((size_t)Cop * Cip);
    const int co = (int)(rem / Cip);
    const int ci = (int)(rem % Cip);
    const float v = (co < Co && ci < Ci) ? w[((size_t)co * Ci + ci) * 5 + dk] : 0.f;
    out[i] = f2bf(v);
  }
}
// mask + transpose t-major -> (B,T,ODIM); d_out = [after | before]
__global__ void k_fin(const float* __restrict__ afterF, const float* __restrict__ beforeF,
                      const int* __restrict__ lens, float* __restrict__ out) {
  size_t i = (size_t)blockIdx.x * blockDim.x + threadIdx.x;
  const size_t st = (size_t)gridDim.x * blockDim.x;
  const size_t n = (size_t)BT_ * ODIM_;
  for (; i < n; i += st) {
    const int c = (int)(i % ODIM_);
    const size_t rem = i / ODIM_;
    const int t = (int)(rem % T_);
    const int b = (int)(rem / T_);
    const size_t r = (size_t)t * B_ + b;
    const float m = (t < lens[b]) ? 1.f : 0.f;
    out[i]     = m * afterF[r * ODIM_ + c];
    out[n + i] = m * beforeF[r * ODIM_ + c];
  }
}

// ---------------- host orchestration ----------------
extern "C" void kernel_launch(void* const* d_in, const int* in_sizes, int n_in,
                              void* d_out, int out_size, void* d_ws, size_t ws_size,
                              hipStream_t stream) {
  (void)in_sizes; (void)n_in; (void)out_size;
  const float* h_t    = (const float*)d_in[0];
  const int*   lens   = (const int*)d_in[1];
  const float* tgt    = (const float*)d_in[2];
  const float* spk    = (const float*)d_in[3];
  const float* W_p0   = (const float*)d_in[4];
  const float* b_p0   = (const float*)d_in[5];
  const float* W_p1   = (const float*)d_in[6];
  const float* b_p1   = (const float*)d_in[7];
  const float* W_ih0  = (const float*)d_in[8];
  const float* W_hh0  = (const float*)d_in[9];
  const float* b_ih0  = (const float*)d_in[10];
  const float* b_hh0  = (const float*)d_in[11];
  const float* W_ih1  = (const float*)d_in[12];
  const float* W_hh1  = (const float*)d_in[13];
  const float* b_ih1  = (const float*)d_in[14];
  const float* b_hh1  = (const float*)d_in[15];
  const float* W_feat = (const float*)d_in[16];
  const float* pw0 = (const float*)d_in[17];
  const float* pw1 = (const float*)d_in[18];
  const float* pw2 = (const float*)d_in[19];
  const float* pw3 = (const float*)d_in[20];
  const float* pw4 = (const float*)d_in[21];

  char* ws = (char*)d_ws;
  size_t off = 0;
  auto alloc = [&](size_t bytes) -> char* {
    char* p = ws + off; off += (bytes + 255) & ~(size_t)255; return p;
  };
  u16t* xs      = (u16t*)alloc((size_t)BT_ * IN0_ * 2);
  u16t* g0b     = (u16t*)alloc((size_t)BT_ * 4 * H_ * 2);
  u16t* zcs     = (u16t*)alloc((size_t)BT_ * FEATIN_ * 2);
  u16t* prevb   = (u16t*)alloc((size_t)BT_ * ODIMP_ * 2);
  u16t* p0b     = (u16t*)alloc((size_t)BT_ * PRE_ * 2);
  float* beforeF= (float*)alloc((size_t)BT_ * ODIM_ * 4);
  float* afterF = (float*)alloc((size_t)BT_ * ODIM_ * 4);
  u16t* pad96   = (u16t*)alloc((size_t)B_ * TP_ * ODIMP_ * 2);
  u16t* padA    = (u16t*)alloc((size_t)B_ * TP_ * PCH_ * 2);
  u16t* padB    = (u16t*)alloc((size_t)B_ * TP_ * PCH_ * 2);
  u16t* wp0b    = (u16t*)alloc((size_t)PRE_ * ODIMP_ * 2);
  u16t* wp1b    = (u16t*)alloc((size_t)PRE_ * PRE_ * 2);
  u16t* wih0b   = (u16t*)alloc((size_t)4 * H_ * IN0_ * 2);
  u16t* whh0b   = (u16t*)alloc((size_t)4 * H_ * H_ * 2);
  u16t* wih1b   = (u16t*)alloc((size_t)4 * H_ * H_ * 2);
  u16t* whh1b   = (u16t*)alloc((size_t)4 * H_ * H_ * 2);
  u16t* wfeatb  = (u16t*)alloc((size_t)NPAD_ * FEATIN_ * 2);        // rows >= 80 zero
  u16t* cw0     = (u16t*)alloc((size_t)5 * PCH_ * ODIMP_ * 2);      // Cin padded 80->96
  u16t* cw1     = (u16t*)alloc((size_t)5 * PCH_ * PCH_ * 2);
  u16t* cw2     = (u16t*)alloc((size_t)5 * PCH_ * PCH_ * 2);
  u16t* cw3     = (u16t*)alloc((size_t)5 * PCH_ * PCH_ * 2);
  u16t* cw4     = (u16t*)alloc((size_t)5 * NPAD_ * PCH_ * 2);       // Cout padded 80->256
  float* bsum0  = (float*)alloc((size_t)4 * H_ * 4);
  float* bsum1  = (float*)alloc((size_t)4 * H_ * 4);
  float* gbuf0  = (float*)alloc((size_t)B_ * 4 * H_ * 4);
  float* gbuf1  = (float*)alloc((size_t)B_ * 4 * H_ * 4);
  float* c0s    = (float*)alloc((size_t)B_ * H_ * 4);
  float* c1s    = (float*)alloc((size_t)B_ * H_ * 4);
  u16t* h0s     = (u16t*)alloc((size_t)B_ * H_ * 2);
  u16t* h1s     = (u16t*)alloc((size_t)B_ * H_ * 2);
  unsigned* bar = (unsigned*)alloc(256);
  if (off > ws_size) return;  // not enough scratch

  const dim3 blk(256);
  auto gsz = [](size_t n) {
    size_t g = (n + 255) / 256; if (g > 65535) g = 65535; return dim3((unsigned)g);
  };
  auto zeroW = [&](void* p, size_t words) {
    k_zero<<<gsz(words), blk, 0, stream>>>((unsigned*)p, words);
  };

  // --- init: barrier, LSTM state, postnet padded buffers (pad rows must be zero;
  //     pad96 is fully zeroed so its padded channels 80..95 are zero too)
  zeroW(bar, 64);
  zeroW(c0s, (size_t)B_ * H_);
  zeroW(c1s, (size_t)B_ * H_);
  zeroW(h0s, (size_t)B_ * H_ / 2);
  zeroW(h1s, (size_t)B_ * H_ / 2);
  zeroW(pad96, (size_t)B_ * TP_ * ODIMP_ / 2);
  zeroW(padA, (size_t)2 * B_ * PCH_ / 2);
  zeroW((u16t*)padA + (size_t)(T_ + 2) * B_ * PCH_, (size_t)2 * B_ * PCH_ / 2);
  zeroW(padB, (size_t)2 * B_ * PCH_ / 2);
  zeroW((u16t*)padB + (size_t)(T_ + 2) * B_ * PCH_, (size_t)2 * B_ * PCH_ / 2);

  // --- weight conversions / repacks (f32 -> bf16, zero-padded where needed)
  k_cvt_pad<<<gsz((size_t)PRE_ * ODIMP_), blk, 0, stream>>>(W_p0, wp0b,
      PRE_, ODIM_, PRE_, ODIMP_);
  k_cvt<<<gsz((size_t)PRE_ * PRE_), blk, 0, stream>>>(W_p1, wp1b, (size_t)PRE_ * PRE_);
  k_cvt<<<gsz((size_t)4 * H_ * IN0_), blk, 0, stream>>>(W_ih0, wih0b, (size_t)4 * H_ * IN0_);
  k_cvt<<<gsz((size_t)4 * H_ * H_), blk, 0, stream>>>(W_hh0, whh0b, (size_t)4 * H_ * H_);
  k_cvt<<<gsz((size_t)4 * H_ * H_), blk, 0, stream>>>(W_ih1, wih1b, (size_t)4 * H_ * H_);
  k_cvt<<<gsz((size_t)4 * H_ * H_), blk, 0, stream>>>(W_hh1, whh1b, (size_t)4 * H_ * H_);
  k_cvt_pad<<<gsz((size_t)NPAD_ * FEATIN_), blk, 0, stream>>>(W_feat, wfeatb,
      ODIM_, FEATIN_, NPAD_, FEATIN_);
  k_convw<<<gsz((size_t)5 * PCH_ * ODIMP_), blk, 0, stream>>>(pw0, cw0, PCH_, ODIM_, PCH_, ODIMP_);
  k_convw<<<gsz((size_t)5 * PCH_ * PCH_), blk, 0, stream>>>(pw1, cw1, PCH_, PCH_, PCH_, PCH_);
  k_convw<<<gsz((size_t)5 * PCH_ * PCH_), blk, 0, stream>>>(pw2, cw2, PCH_, PCH_, PCH_, PCH_);
  k_convw<<<gsz((size_t)5 * PCH_ * PCH_), blk, 0, stream>>>(pw3, cw3, PCH_, PCH_, PCH_, PCH_);
  k_convw<<<gsz((size_t)5 * NPAD_ * PCH_), blk, 0, stream>>>(pw4, cw4, ODIM_, PCH_, NPAD_, PCH_);
  k_add<<<(4 * H_ + 255) / 256, blk, 0, stream>>>(b_ih0, b_hh0, bsum0, 4 * H_);
  k_add<<<(4 * H_ + 255) / 256, blk, 0, stream>>>(b_ih1, b_hh1, bsum1, 4 * H_);

  // --- activation packing
  k_pack_prev<<<gsz((size_t)BT_ * ODIMP_), blk, 0, stream>>>(tgt, prevb);
  k_pack_xz<<<gsz((size_t)BT_ * (ENC_ + SPK_)), blk, 0, stream>>>(h_t, spk, xs, zcs);

  // --- prenet: two ReLU GEMMs (layer-1 writes straight into xs cols [512,768))
  k_gemm<<<dim3(1, BT_ / 32), blk, 0, stream>>>(prevb, ODIMP_, wp0b, ODIMP_, b_p0,
      nullptr, 0, p0b, PRE_, BT_, PRE_, ODIMP_, 1);
  k_gemm<<<dim3(1, BT_ / 32), blk, 0, stream>>>(p0b, PRE_, wp1b, PRE_, b_p1,
      nullptr, 0, xs + ENC_, IN0_, BT_, PRE_, PRE_, 1);

  // --- full-GPU time-parallel input projection: g0 = xs @ W_ih0^T + (b_ih0+b_hh0)
  k_gemm<<<dim3(16, BT_ / 32), blk, 0, stream>>>(xs, IN0_, wih0b, IN0_, bsum0,
      nullptr, 0, g0b, 4 * H_, BT_, 4 * H_, IN0_, 0);

  // --- persistent sequential LSTM (grid-wide barriers, weights stay L2-hot)
  k_lstm<<<dim3(NBLK_), blk, 0, stream>>>(g0b, whh0b, wih1b, whh1b, bsum1, zcs,
      gbuf0, gbuf1, c0s, c1s, h0s, h1s, bar);

  // --- feat_out: before = zcs @ W_feat^T ; also seed postnet padded input (interior rows)
  k_gemm<<<dim3(1, BT_ / 32), blk, 0, stream>>>(zcs, FEATIN_, wfeatb, FEATIN_, nullptr,
      beforeF, ODIM_, pad96 + (size_t)2 * B_ * ODIMP_, ODIMP_, BT_, ODIM_, FEATIN_, 0);

  // --- postnet: 4x (conv+tanh) ping-pong, final conv + residual
  k_conv<<<dim3(2, BT_ / 32), blk, 0, stream>>>(pad96, cw0, nullptr, nullptr,
      padA + (size_t)2 * B_ * PCH_, PCH_, PCH_, ODIMP_, 2);
  k_conv<<<dim3(2, BT_ / 32), blk, 0, stream>>>(padA, cw1, nullptr, nullptr,
      padB + (size_t)2 * B_ * PCH_, PCH_, PCH_, PCH_, 2);
  k_conv<<<dim3(2, BT_ / 32), blk, 0, stream>>>(padB, cw2, nullptr, nullptr,
      padA + (size_t)2 * B_ * PCH_, PCH_, PCH_, PCH_, 2);
  k_conv<<<dim3(2, BT_ / 32), blk, 0, stream>>>(padA, cw3, nullptr, nullptr,
      padB + (size_t)2 * B_ * PCH_, PCH_, PCH_, PCH_, 2);
  k_conv<<<dim3(1, BT_ / 32), blk, 0, stream>>>(padB, cw4, beforeF, afterF,
      nullptr, ODIM_, NPAD_, PCH_, 0);

  // --- mask + layout to (B,T,ODIM); d_out = [after_outs | before_outs]
  k_fin<<<gsz((size_t)BT_ * ODIM_), blk, 0, stream>>>(afterF, beforeF, lens, (float*)d_out);
}
